// PatchedDbrxExperts_29240137351600
// MI455X (gfx1250) — compile-verified, matrix operand order checked by reading, not run on hardware
//
#include <hip/hip_runtime.h>

typedef __attribute__((ext_vector_type(16))) __bf16 v16bf;
typedef __attribute__((ext_vector_type(8)))  float  v8f;

static constexpr int TOKENS = 4096;
static constexpr int HID    = 2048;
static constexpr int INTR   = 4096;
static constexpr int NEXP   = 8;
static constexpr int TOPK   = 4;

static constexpr int BM   = 128;  // token rows per block
static constexpr int BN   = 64;   // output cols per block
static constexpr int KC   = 32;   // K chunk (bf16 WMMA depth)
static constexpr int LDSP = 40;   // padded LDS row stride in ushorts (32 + 8)

struct BF16Frag { uint4 lo, hi; };

// fp32 -> bf16 round-to-nearest-even
__device__ __forceinline__ unsigned short f2bf(float f) {
  unsigned u = __float_as_uint(f);
  u += 0x7FFFu + ((u >> 16) & 1u);
  return (unsigned short)(u >> 16);
}

// Async DMA: 16 bytes global -> LDS, tracked by ASYNCcnt (no VGPR round trip).
__device__ __forceinline__ void async_copy16(const void* gbase, unsigned byte_off,
                                             const unsigned short* lds_dst) {
  unsigned lds_addr = (unsigned)(unsigned long long)lds_dst;  // low 32 bits = LDS offset
  unsigned long long base = (unsigned long long)gbase;
  asm volatile("global_load_async_to_lds_b128 %0, %1, %2"
               :
               : "v"(lds_addr), "v"(byte_off), "s"(base)
               : "memory");
}
__device__ __forceinline__ void wait_async() {
  asm volatile("s_wait_asynccnt 0x0" ::: "memory");
}

// Load one 16x32 bf16 WMMA operand fragment from LDS (row-major [row][K], padded).
// CDNA5 16-bit layout: lane L holds row (L&15); lanes 0-15 take K 0..7 / 16..23,
// lanes 16-31 take K 8..15 / 24..31.
__device__ __forceinline__ v16bf load_frag(const unsigned short* lds, int row0) {
  const int lane = threadIdx.x & 31;
  const int r    = row0 + (lane & 15);
  const int kh   = (lane >> 4) << 3;
  BF16Frag t;
  t.lo = *reinterpret_cast<const uint4*>(lds + r * LDSP + kh);
  t.hi = *reinterpret_cast<const uint4*>(lds + r * LDSP + kh + 16);
  return __builtin_bit_cast(v16bf, t);
}

// ---- Kernel 1: act_e = coef_e * silu(x @ Wg^T) * (x @ Wu^T), bf16 out ----
// expert = e_base + blockIdx.z ; act slot = blockIdx.z
__global__ __launch_bounds__(256) void moe_gate_up(
    const float* __restrict__ x,            // [T, HID]
    const float* __restrict__ wgu_all,      // [E, 2*INTR, HID]
    const float* __restrict__ top_w,        // [T, TOPK]
    const int*   __restrict__ top_e,        // [T, TOPK]
    unsigned short* __restrict__ act_all,   // [nslots, T, INTR] bf16
    int e_base)
{
  __shared__ __align__(16) unsigned short sA[BM * LDSP];
  __shared__ __align__(16) unsigned short sB[2 * BN * LDSP];
  __shared__ float sCoef[BM];

  const int expert = e_base + blockIdx.z;
  const float* wgu = wgu_all + (size_t)expert * 2 * INTR * HID;
  unsigned short* act = act_all + (size_t)blockIdx.z * TOKENS * INTR;

  const int t0  = blockIdx.y * BM;
  const int n0  = blockIdx.x * BN;
  const int tid = threadIdx.x;
  const int wid = tid >> 5;
  const int wm  = wid >> 1;   // 0..3
  const int wn  = wid & 1;    // 0..1

  if (tid < BM) {
    int t = t0 + tid;
    float c = 0.0f;
#pragma unroll
    for (int k = 0; k < TOPK; ++k)
      c += (top_e[t * TOPK + k] == expert) ? top_w[t * TOPK + k] : 0.0f;
    sCoef[tid] = c;
  }

  v8f zero = {};
  v8f accg[2][2], accu[2][2];
#pragma unroll
  for (int i = 0; i < 2; ++i)
#pragma unroll
    for (int j = 0; j < 2; ++j) { accg[i][j] = zero; accu[i][j] = zero; }

  for (int k0 = 0; k0 < HID; k0 += KC) {
    __syncthreads();
    // x tile: 128 rows x 8 float4, 4 per thread; convert fp32->bf16 into LDS
#pragma unroll
    for (int i = 0; i < 4; ++i) {
      int idx = tid + i * 256;
      int r = idx >> 3, c4 = idx & 7;
      const float* p = x + (size_t)(t0 + r) * HID + k0 + c4 * 4;
      float4 v = *reinterpret_cast<const float4*>(p);
      unsigned short* d = sA + r * LDSP + c4 * 4;
      d[0] = f2bf(v.x); d[1] = f2bf(v.y); d[2] = f2bf(v.z); d[3] = f2bf(v.w);
    }
    // weight tile: rows 0..63 gate, 64..127 up
#pragma unroll
    for (int i = 0; i < 4; ++i) {
      int idx = tid + i * 256;
      int r = idx >> 3, c4 = idx & 7;
      int wrow = (r < BN) ? (n0 + r) : (INTR + n0 + (r - BN));
      const float* p = wgu + (size_t)wrow * HID + k0 + c4 * 4;
      float4 v = *reinterpret_cast<const float4*>(p);
      if (k0 + KC < HID) __builtin_prefetch(p + KC, 0, 1);
      unsigned short* d = sB + r * LDSP + c4 * 4;
      d[0] = f2bf(v.x); d[1] = f2bf(v.y); d[2] = f2bf(v.z); d[3] = f2bf(v.w);
    }
    __syncthreads();

    v16bf a[2], bg[2], bu[2];
    a[0]  = load_frag(sA, wm * 32);
    a[1]  = load_frag(sA, wm * 32 + 16);
    bg[0] = load_frag(sB, wn * 32);
    bg[1] = load_frag(sB, wn * 32 + 16);
    bu[0] = load_frag(sB, BN + wn * 32);
    bu[1] = load_frag(sB, BN + wn * 32 + 16);

#pragma unroll
    for (int sm = 0; sm < 2; ++sm)
#pragma unroll
      for (int sn = 0; sn < 2; ++sn) {
        accg[sm][sn] = __builtin_amdgcn_wmma_f32_16x16x32_bf16(
            false, a[sm], false, bg[sn], (short)0, accg[sm][sn], false, false);
        accu[sm][sn] = __builtin_amdgcn_wmma_f32_16x16x32_bf16(
            false, a[sm], false, bu[sn], (short)0, accu[sm][sn], false, false);
      }
  }

  // Epilogue: coef * silu(gate) * up -> bf16. C layout: VGPR i -> M=i / 8+i by lane half.
  const int lane = tid & 31;
  const int nl   = lane & 15;
  const int mh   = (lane >> 4) * 8;
#pragma unroll
  for (int sm = 0; sm < 2; ++sm)
#pragma unroll
    for (int sn = 0; sn < 2; ++sn)
#pragma unroll
      for (int i = 0; i < 8; ++i) {
        int   rl = wm * 32 + sm * 16 + mh + i;
        float g  = accg[sm][sn][i];
        float u  = accu[sm][sn][i];
        float sig = __builtin_amdgcn_rcpf(1.0f + __expf(-g));  // fast sigmoid
        float s  = sCoef[rl] * g * sig * u;
        int t = t0 + rl;
        int n = n0 + wn * 32 + sn * 16 + nl;
        act[(size_t)t * INTR + n] = f2bf(s);
      }
}

// ---- Kernel 2: out(+)= sum_{e in [e0,e1)} act_e @ Wd_e^T  (act pre-scaled by coef) ----
__global__ __launch_bounds__(256) void moe_down(
    const unsigned short* __restrict__ act,  // slot for e0: [T, INTR] bf16
    size_t act_stride,                       // elements between expert slots
    const float* __restrict__ wd_all,        // [E, HID, INTR]
    int e0, int e1,
    float* __restrict__ out,                 // [T, HID]
    int accumulate)
{
  __shared__ __align__(16) unsigned short sA[BM * LDSP];
  __shared__ __align__(16) unsigned short sB[BN * LDSP];

  const int t0  = blockIdx.y * BM;
  const int d0  = blockIdx.x * BN;
  const int tid = threadIdx.x;
  const int wid = tid >> 5;
  const int wm  = wid >> 1;
  const int wn  = wid & 1;

  v8f zero = {};
  v8f acc[2][2];
#pragma unroll
  for (int i = 0; i < 2; ++i)
#pragma unroll
    for (int j = 0; j < 2; ++j) acc[i][j] = zero;

  for (int e = e0; e < e1; ++e) {
    const unsigned short* actp = act + (size_t)(e - e0) * act_stride;
    const float* wd = wd_all + (size_t)e * HID * INTR;

    for (int k0 = 0; k0 < INTR; k0 += KC) {
      __syncthreads();
      // act tile is already bf16: async DMA global->LDS (128 rows x 4 x 16B, 2/thread)
#pragma unroll
      for (int i = 0; i < 2; ++i) {
        int idx = tid + i * 256;
        int r = idx >> 2, c4 = idx & 3;
        unsigned off = (unsigned)(((size_t)(t0 + r) * INTR + k0 + c4 * 8) * 2);
        async_copy16(actp, off, sA + r * LDSP + c4 * 8);
      }
      // down-weight tile: 64 rows x 8 float4, 2/thread; convert fp32->bf16
#pragma unroll
      for (int i = 0; i < 2; ++i) {
        int idx = tid + i * 256;
        int r = idx >> 3, c4 = idx & 7;
        const float* p = wd + (size_t)(d0 + r) * INTR + k0 + c4 * 4;
        float4 v = *reinterpret_cast<const float4*>(p);
        if (k0 + KC < INTR) __builtin_prefetch(p + KC, 0, 1);
        unsigned short* d = sB + r * LDSP + c4 * 4;
        d[0] = f2bf(v.x); d[1] = f2bf(v.y); d[2] = f2bf(v.z); d[3] = f2bf(v.w);
      }
      wait_async();
      __syncthreads();

      v16bf a[2], b[2];
      a[0] = load_frag(sA, wm * 32);
      a[1] = load_frag(sA, wm * 32 + 16);
      b[0] = load_frag(sB, wn * 32);
      b[1] = load_frag(sB, wn * 32 + 16);

#pragma unroll
      for (int sm = 0; sm < 2; ++sm)
#pragma unroll
        for (int sn = 0; sn < 2; ++sn)
          acc[sm][sn] = __builtin_amdgcn_wmma_f32_16x16x32_bf16(
              false, a[sm], false, b[sn], (short)0, acc[sm][sn], false, false);
    }
  }

  const int lane = tid & 31;
  const int nl   = lane & 15;
  const int mh   = (lane >> 4) * 8;
#pragma unroll
  for (int sm = 0; sm < 2; ++sm)
#pragma unroll
    for (int sn = 0; sn < 2; ++sn)
#pragma unroll
      for (int i = 0; i < 8; ++i) {
        int t = t0 + wm * 32 + sm * 16 + mh + i;
        int d = d0 + wn * 32 + sn * 16 + nl;
        size_t o = (size_t)t * HID + d;
        float v = acc[sm][sn][i];
        if (accumulate) out[o] += v; else out[o] = v;
      }
}

extern "C" void kernel_launch(void* const* d_in, const int* in_sizes, int n_in,
                              void* d_out, int out_size, void* d_ws, size_t ws_size,
                              hipStream_t stream) {
  const float* x     = (const float*)d_in[0];
  // d_in[1] = full router weights (unused by the patched forward)
  const float* top_w = (const float*)d_in[2];
  const int*   top_e = (const int*)d_in[3];
  const float* wgu   = (const float*)d_in[4];   // [8, 2*INTR, HID]
  const float* wdn   = (const float*)d_in[5];   // [8, HID, INTR]
  float* out = (float*)d_out;
  unsigned short* act = (unsigned short*)d_ws;

  const size_t per_expert = (size_t)TOKENS * INTR;          // bf16 elements per expert
  const bool   big = ws_size >= per_expert * NEXP * sizeof(unsigned short);

  dim3 blk(256);
  if (big) {
    // All experts' act at once (268 MB ws), output written exactly once.
    dim3 g1(INTR / BN, TOKENS / BM, NEXP);
    dim3 g2(HID  / BN, TOKENS / BM);
    moe_gate_up<<<g1, blk, 0, stream>>>(x, wgu, top_w, top_e, act, 0);
    moe_down<<<g2, blk, 0, stream>>>(act, per_expert, wdn, 0, NEXP, out, 0);
  } else {
    // Fallback: stage one expert at a time (33.5 MB ws), accumulate output.
    dim3 g1(INTR / BN, TOKENS / BM, 1);
    dim3 g2(HID  / BN, TOKENS / BM);
    for (int e = 0; e < NEXP; ++e) {
      moe_gate_up<<<g1, blk, 0, stream>>>(x, wgu, top_w, top_e, act, e);
      moe_down<<<g2, blk, 0, stream>>>(act, 0, wdn, e, e + 1, out, e != 0);
    }
  }
}
